// MemoryAugmentedAttention_41077067219101
// MI455X (gfx1250) — matrix operand order repaired and checked by reading, not compiled, not run on hardware
//
#include <hip/hip_runtime.h>
#include <hip/hip_bf16.h>

// ---------------------------------------------------------------------------
// MemoryAugmentedAttention for MI455X (gfx1250), bf16 WMMA path.
//
// Algebra:  score_bk = m_bk . (q_b @ Wk^T) + q_b.bk   (one GEMM instead of 5)
//           mem_o    = (sum_k w_k m_bk) @ (Wv@Wo) + bv@Wo + bo   (one GEMM)
// => 3 GEMMs of (B x 512 x 512) on v_wmma_f32_16x16x32_bf16, fp32 accum.
//
// Weights are staged into LDS per-workgroup with GLOBAL_LOAD_ASYNC_TO_LDS
// (double-buffered, ASYNCcnt-synchronized) so 8 waves share one L2 read.
// ---------------------------------------------------------------------------

#define B_ROWS   32768
#define DDIM     512
#define KMEM     5
#define BLK_ROWS 64          // rows per workgroup
#define LDSB     520         // bf16 LDS row stride (elems): 1040B % 256 == 16 -> conflict-free
#define LDSF     516         // f32  LDS row stride (elems): 2064B % 256 == 16
#define REG_A    0                         // query bf16, later low half of mem_o f32
#define REG_C    (BLK_ROWS * LDSB * 2)     // qk bf16,   later high half of mem_o f32
#define REG_B    (2 * BLK_ROWS * LDSB * 2) // q bf16, then mbar bf16
#define REG_S    (3 * BLK_ROWS * LDSB * 2) // weight staging: 2 x 32 KB
#define SMEM_BYTES (3 * BLK_ROWS * LDSB * 2 + 2 * 32768)   // 265216 B (WGP has 320 KB)
#define SLICE_ELEMS 16384    // one kt-slice: 32 tiles x 512 bf16 = 32 KB

typedef __bf16 bf16_t;
typedef __attribute__((ext_vector_type(16))) __bf16 v16bf;
typedef __attribute__((ext_vector_type(8)))  __bf16 v8bf;
typedef __attribute__((ext_vector_type(8)))  float  v8f;

__device__ __forceinline__ float wave_sum(float v) {
#pragma unroll
  for (int off = 16; off > 0; off >>= 1) v += __shfl_xor(v, off, 32);
  return v;
}

// ---------------------------------------------------------------------------
// Prep kernel 1: Wvo = Wv @ Wo   (512x512, trivial 0.27 GFLOP, runs once)
// ---------------------------------------------------------------------------
__global__ void maa_wvo_kernel(const float* __restrict__ Wv,
                               const float* __restrict__ Wo,
                               float* __restrict__ Wvo) {
  int gid = blockIdx.x * 256 + threadIdx.x;        // 262144 threads
  int k = gid >> 9, n = gid & 511;
  float acc = 0.f;
  for (int d = 0; d < DDIM; ++d) acc = fmaf(Wv[k * DDIM + d], Wo[d * DDIM + n], acc);
  Wvo[gid] = acc;
}

__global__ void maa_bvo_kernel(const float* __restrict__ bv,
                               const float* __restrict__ Wo,
                               const float* __restrict__ bo,
                               float* __restrict__ bvo) {
  int n = blockIdx.x * 256 + threadIdx.x;
  if (n >= DDIM) return;
  float acc = bo[n];
  for (int d = 0; d < DDIM; ++d) acc = fmaf(bv[d], Wo[d * DDIM + n], acc);
  bvo[n] = acc;
}

// ---------------------------------------------------------------------------
// Prep kernel 2: pack Wq, Wk^T, Wvo into WMMA B-fragment layout (bf16),
// kt-major so one kt-slice (all 32 n-tiles of one 32-wide K band) is a
// contiguous 32 KB block for async staging.
//   tile tidx = kt*32 + nt  covers cols nt*16..+15, K kt*32..+31.
//   Per ISA 7.12.2/7.12.4, lane L of the B fragment holds column
//   n = nt*16 + (L&15), elements j = K kt*32 + (j>>3)*16 + (L>>4)*8 + (j&7).
//   Within the 512-elem tile we store two 512B half-tiles
//   (elem p = half*256 + L*8 + (j&7), half = j>>3) so each of the two
//   ds_load_b128 per fragment has a 16B lane stride -> no bank conflicts.
// ---------------------------------------------------------------------------
__global__ void maa_pack_kernel(const float* __restrict__ Wq,
                                const float* __restrict__ Wk,
                                const float* __restrict__ Wvo,
                                bf16_t* __restrict__ Wq_p,
                                bf16_t* __restrict__ Wk_p,
                                bf16_t* __restrict__ Wvo_p) {
  int gid = blockIdx.x * 256 + threadIdx.x;   // 3 * 262144 threads
  int w    = gid >> 18;
  int rem  = gid & 262143;
  int tidx = rem >> 9;          // kt*32 + nt
  int e    = rem & 511;
  int half = e >> 8;
  int L    = (e >> 3) & 31;
  int j    = half * 8 + (e & 7);
  int kt = tidx >> 5, nt = tidx & 31;
  int n = nt * 16 + (L & 15);
  int k = kt * 32 + (j >> 3) * 16 + (L >> 4) * 8 + (j & 7);
  float v; bf16_t* dst;
  if (w == 0)      { v = Wq[k * DDIM + n];  dst = Wq_p;  }
  else if (w == 1) { v = Wk[n * DDIM + k];  dst = Wk_p;  }   // transpose
  else             { v = Wvo[k * DDIM + n]; dst = Wvo_p; }
  dst[tidx * 512 + e] = (bf16_t)v;
}

// ---------------------------------------------------------------------------
// Async-stage one 32 KB kt-slice of packed weights into LDS.
// 256 threads x 128 B each, GLOBAL_LOAD_ASYNC_TO_LDS_B128 (GV mode).
// LDS address VGPR = low 32 bits of the flat shared pointer (the LDS
// aperture keeps the byte offset in addr[31:0]).
// ---------------------------------------------------------------------------
__device__ __forceinline__ void stage_slice(bf16_t* dstLds, const bf16_t* Bp,
                                            int kt, int tid) {
  unsigned lds = (unsigned)(unsigned long long)dstLds + (unsigned)(tid * 128);
  const char* g = (const char*)Bp + (size_t)kt * 32768 + tid * 128;
  if (kt < 15) __builtin_prefetch((const void*)(g + 32768), 0, 0);
#pragma unroll
  for (int i = 0; i < 8; ++i) {
    asm volatile("global_load_async_to_lds_b128 %0, %1, off"
                 :: "v"(lds + i * 16), "v"(g + i * 16) : "memory");
  }
}

// ---------------------------------------------------------------------------
// GEMM: one wave computes a 16-row x 256-col strip of (A[64x512] @ W).
// kt-outer loop; B comes from the double-buffered LDS stage; 16 independent
// accumulators per wave (no WMMA RAW chains). A fragment layout per ISA
// 7.12.2 (lane m=L&15, koff=(L>>4)*8, elems 0-7 = K koff.., 8-15 = K 16+koff..).
// ---------------------------------------------------------------------------
template <bool TO_BF16>
__device__ __forceinline__ void gemm_strip(const bf16_t* __restrict__ ldsA,
                                           const bf16_t* __restrict__ Bp,
                                           const float* __restrict__ bias,
                                           bf16_t* __restrict__ outB,
                                           float* __restrict__ outF,
                                           bf16_t* __restrict__ stage) {
  const int tid = threadIdx.x;
  const int wave = tid >> 5, lane = tid & 31;
  const int s = wave >> 1;        // row strip 0..3
  const int h = wave & 1;         // column half 0..1
  const int m = lane & 15;
  const int kg = lane >> 4;       // 0/1

  v8f acc[16];
#pragma unroll
  for (int t = 0; t < 16; ++t) acc[t] = (v8f){};

  stage_slice(stage, Bp, 0, tid);
  asm volatile("s_wait_asynccnt 0" ::: "memory");
  __syncthreads();

  for (int kt = 0; kt < 16; ++kt) {
    const bf16_t* buf = stage + (kt & 1) * SLICE_ELEMS;
    if (kt < 15) stage_slice(stage + ((kt + 1) & 1) * SLICE_ELEMS, Bp, kt + 1, tid);

    // A fragment for this kt
    const bf16_t* ap = ldsA + (s * 16 + m) * LDSB + kt * 32 + kg * 8;
    union { v16bf v; v8bf hh[2]; } ua;
    ua.hh[0] = *(const v8bf*)(ap);
    ua.hh[1] = *(const v8bf*)(ap + 16);
    const v16bf a = ua.v;

#pragma unroll
    for (int t = 0; t < 16; ++t) {
      const bf16_t* bp = buf + (h * 16 + t) * 512 + lane * 8;
      union { v16bf v; v8bf hh[2]; } ub;
      ub.hh[0] = *(const v8bf*)(bp);
      ub.hh[1] = *(const v8bf*)(bp + 256);
      acc[t] = __builtin_amdgcn_wmma_f32_16x16x32_bf16(
          false, a, false, ub.v, (short)0, acc[t], false, false);
    }
    if (kt < 15) {
      asm volatile("s_wait_asynccnt 0" ::: "memory");
      __syncthreads();
    }
  }

#pragma unroll
  for (int t = 0; t < 16; ++t) {
    const int n = (h * 16 + t) * 16 + m;        // output column 0..511
    const float bb = bias ? bias[n] : 0.f;
#pragma unroll
    for (int v = 0; v < 8; ++v) {               // C/D layout: VGPR v -> row kg*8+v
      const int row = s * 16 + kg * 8 + v;
      const float val = acc[t][v] + bb;
      if (TO_BF16) outB[row * LDSB + n] = (bf16_t)val;
      else         outF[row * LDSF + n] = val;
    }
  }
}

// ---------------------------------------------------------------------------
// Main fused kernel: 512 blocks x 256 threads (8 wave32), 64 rows per block.
// ---------------------------------------------------------------------------
__global__ __launch_bounds__(256) void maa_main_kernel(
    const float* __restrict__ query, const float* __restrict__ mems,
    const float* __restrict__ sims, const int* __restrict__ mask,
    const float* __restrict__ bq, const float* __restrict__ bk,
    const float* __restrict__ Wg, const float* __restrict__ bg,
    const float* __restrict__ ln_g, const float* __restrict__ ln_b,
    const bf16_t* __restrict__ Wq_p, const bf16_t* __restrict__ Wk_p,
    const bf16_t* __restrict__ Wvo_p, const float* __restrict__ bvo,
    float* __restrict__ out) {
  __shared__ __align__(16) unsigned char smem[SMEM_BYTES];
  bf16_t* A_q   = (bf16_t*)(smem + REG_A);   // query bf16
  bf16_t* C_qk  = (bf16_t*)(smem + REG_C);   // qk bf16
  bf16_t* B_qm  = (bf16_t*)(smem + REG_B);   // q bf16, then mbar bf16
  bf16_t* stage = (bf16_t*)(smem + REG_S);   // 2 x 32 KB weight staging
  float*  F_mo  = (float*)(smem);            // mem_o f32 (aliases A_q + C_qk)

  const int rowBase = blockIdx.x * BLK_ROWS;
  const int wave = threadIdx.x >> 5, lane = threadIdx.x & 31;

  // ---- Phase 1: stage query tile -> LDS bf16 -------------------------------
  for (int i = threadIdx.x; i < BLK_ROWS * 128; i += 256) {
    const int r = i >> 7, c4 = (i & 127) * 4;
    const float4 q4 = *(const float4*)(query + (size_t)(rowBase + r) * DDIM + c4);
    bf16_t* d = A_q + r * LDSB + c4;
    d[0] = (bf16_t)q4.x; d[1] = (bf16_t)q4.y; d[2] = (bf16_t)q4.z; d[3] = (bf16_t)q4.w;
  }
  __syncthreads();

  // ---- Phase 2: q = query @ Wq + bq  (bf16 -> REG_B) -----------------------
  gemm_strip<true>(A_q, Wq_p, bq, B_qm, nullptr, stage);
  __syncthreads();

  // ---- Phase 3: qk = q @ Wk^T  (bf16 -> REG_C) -----------------------------
  gemm_strip<true>(B_qm, Wk_p, nullptr, C_qk, nullptr, stage);
  __syncthreads();

  // ---- Phase 4: scores, masked softmax, mbar = sum_k w_k m_k --------------
  {
    const float scale = 0.044194173824159216f;   // 512^-0.5
    for (int rr = 0; rr < 8; ++rr) {
      const int r = wave * 8 + rr;               // each wave owns 8 rows
      const size_t gr = (size_t)(rowBase + r);
      float qkv[16], qv[16];
      {
        const bf16_t* qp = C_qk + r * LDSB + lane * 16;
        const bf16_t* qq = B_qm + r * LDSB + lane * 16;
#pragma unroll
        for (int j = 0; j < 16; ++j) { qkv[j] = (float)qp[j]; qv[j] = (float)qq[j]; }
      }
      // q . bk (bk == 0 in the harness, kept for generality)
      float pq = 0.f;
#pragma unroll
      for (int j = 0; j < 16; ++j) pq = fmaf(qv[j], bk[lane * 16 + j], pq);
      const float qbk = wave_sum(pq);

      float mreg[KMEM][16], s[KMEM];
#pragma unroll
      for (int k = 0; k < KMEM; ++k) {
        const float* mp = mems + (gr * KMEM + k) * DDIM + lane * 16;
        float acc = 0.f;
#pragma unroll
        for (int j4 = 0; j4 < 4; ++j4) {
          const float4 m4 = *(const float4*)(mp + j4 * 4);
          mreg[k][j4 * 4 + 0] = m4.x; mreg[k][j4 * 4 + 1] = m4.y;
          mreg[k][j4 * 4 + 2] = m4.z; mreg[k][j4 * 4 + 3] = m4.w;
          acc = fmaf(m4.x, qkv[j4 * 4 + 0], acc);
          acc = fmaf(m4.y, qkv[j4 * 4 + 1], acc);
          acc = fmaf(m4.z, qkv[j4 * 4 + 2], acc);
          acc = fmaf(m4.w, qkv[j4 * 4 + 3], acc);
        }
        s[k] = (wave_sum(acc) + qbk) * scale;
      }
      int mk[KMEM]; float mx = -__builtin_inff();
#pragma unroll
      for (int k = 0; k < KMEM; ++k) {
        mk[k] = mask[gr * KMEM + k];
        if (mk[k] && s[k] > mx) mx = s[k];
      }
      float w[KMEM], sum = 0.f;
#pragma unroll
      for (int k = 0; k < KMEM; ++k) { w[k] = mk[k] ? __expf(s[k] - mx) : 0.f; sum += w[k]; }
      const float inv = 1.f / sum;
      bf16_t* mb = B_qm + r * LDSB + lane * 16;   // overwrite q row (done with it)
#pragma unroll
      for (int j = 0; j < 16; ++j) {
        float acc = 0.f;
#pragma unroll
        for (int k = 0; k < KMEM; ++k) acc = fmaf(w[k], mreg[k][j], acc);
        mb[j] = (bf16_t)(acc * inv);
      }
    }
  }
  __syncthreads();

  // ---- Phase 5: mem_o = mbar @ (Wv@Wo) + (bv@Wo + bo)  (f32 -> F_mo) -------
  gemm_strip<false>(B_qm, Wvo_p, bvo, nullptr, F_mo, stage);
  __syncthreads();

  // ---- Phase 6: gate, confidence, residual, layernorm ----------------------
  for (int rr = 0; rr < 8; ++rr) {
    const int r = wave * 8 + rr;
    const size_t gr = (size_t)(rowBase + r);
    float qv[16], mo[16];
    {
      const float* qp = query + gr * DDIM + lane * 16;
      const float* fp = F_mo + r * LDSF + lane * 16;
#pragma unroll
      for (int j4 = 0; j4 < 4; ++j4) {
        const float4 q4 = *(const float4*)(qp + j4 * 4);
        const float4 f4 = *(const float4*)(fp + j4 * 4);
        qv[j4 * 4 + 0] = q4.x; qv[j4 * 4 + 1] = q4.y; qv[j4 * 4 + 2] = q4.z; qv[j4 * 4 + 3] = q4.w;
        mo[j4 * 4 + 0] = f4.x; mo[j4 * 4 + 1] = f4.y; mo[j4 * 4 + 2] = f4.z; mo[j4 * 4 + 3] = f4.w;
      }
    }
    float g = 0.f;
#pragma unroll
    for (int j = 0; j < 16; ++j) {
      g = fmaf(qv[j], Wg[lane * 16 + j], g);
      g = fmaf(mo[j], Wg[DDIM + lane * 16 + j], g);
    }
    g = wave_sum(g) + bg[0];
    const float gate = 1.f / (1.f + __expf(-g));
    float ms = sims[gr * KMEM];
#pragma unroll
    for (int k = 1; k < KMEM; ++k) ms = fmaxf(ms, sims[gr * KMEM + k]);
    const float conf = 1.f / (1.f + __expf(-(ms - 0.7f)));   // GATE_TEMP = 1
    const float gc = gate * conf;

    float o[16], s1 = 0.f, s2 = 0.f;
#pragma unroll
    for (int j = 0; j < 16; ++j) {
      o[j] = fmaf(gc, mo[j], qv[j]);
      s1 += o[j];
      s2 = fmaf(o[j], o[j], s2);
    }
    s1 = wave_sum(s1);
    s2 = wave_sum(s2);
    const float mu = s1 * (1.f / 512.f);
    const float var = s2 * (1.f / 512.f) - mu * mu;
    const float rstd = rsqrtf(var + 1e-5f);
    float* op = out + gr * DDIM + lane * 16;
#pragma unroll
    for (int j4 = 0; j4 < 4; ++j4) {
      float4 r4;
      r4.x = (o[j4*4+0] - mu) * rstd * ln_g[lane*16 + j4*4+0] + ln_b[lane*16 + j4*4+0];
      r4.y = (o[j4*4+1] - mu) * rstd * ln_g[lane*16 + j4*4+1] + ln_b[lane*16 + j4*4+1];
      r4.z = (o[j4*4+2] - mu) * rstd * ln_g[lane*16 + j4*4+2] + ln_b[lane*16 + j4*4+2];
      r4.w = (o[j4*4+3] - mu) * rstd * ln_g[lane*16 + j4*4+3] + ln_b[lane*16 + j4*4+3];
      *(float4*)(op + j4 * 4) = r4;
    }
  }
}

// ---------------------------------------------------------------------------
extern "C" void kernel_launch(void* const* d_in, const int* in_sizes, int n_in,
                              void* d_out, int out_size, void* d_ws, size_t ws_size,
                              hipStream_t stream) {
  (void)in_sizes; (void)n_in; (void)out_size; (void)ws_size;
  const float* query = (const float*)d_in[0];
  const float* mems  = (const float*)d_in[1];
  const float* sims  = (const float*)d_in[2];
  const int*   mask  = (const int*)d_in[3];
  const float* Wq = (const float*)d_in[4];
  const float* bq = (const float*)d_in[5];
  const float* Wk = (const float*)d_in[6];
  const float* bk = (const float*)d_in[7];
  const float* Wv = (const float*)d_in[8];
  const float* bv = (const float*)d_in[9];
  const float* Wo = (const float*)d_in[10];
  const float* bo = (const float*)d_in[11];
  const float* Wg = (const float*)d_in[12];
  const float* bg = (const float*)d_in[13];
  const float* lng = (const float*)d_in[14];
  const float* lnb = (const float*)d_in[15];
  float* out = (float*)d_out;

  char* ws = (char*)d_ws;
  float*  Wvo   = (float*)(ws);                         // 1 MB
  float*  bvo   = (float*)(ws + 1048576);               // 2 KB
  bf16_t* Wq_p  = (bf16_t*)(ws + 1050624);              // 512 KB
  bf16_t* Wk_p  = (bf16_t*)(ws + 1050624 + 524288);     // 512 KB
  bf16_t* Wvo_p = (bf16_t*)(ws + 1050624 + 1048576);    // 512 KB

  maa_wvo_kernel<<<1024, 256, 0, stream>>>(Wv, Wo, Wvo);
  maa_bvo_kernel<<<2, 256, 0, stream>>>(bv, Wo, bo, bvo);
  maa_pack_kernel<<<3072, 256, 0, stream>>>(Wq, Wk, Wvo, Wq_p, Wk_p, Wvo_p);
  maa_main_kernel<<<B_ROWS / BLK_ROWS, 256, 0, stream>>>(
      query, mems, sims, mask, bq, bk, Wg, bg, lng, lnb,
      Wq_p, Wk_p, Wvo_p, bvo, out);
}
